// STSLinearAttention_73349451481747
// MI455X (gfx1250) — matrix operand order repaired and checked by reading, not compiled, not run on hardware
//
#include <hip/hip_runtime.h>
#include <hip/hip_bf16.h>

// ---------------------------------------------------------------------------
// Linear attention (elu+1 feature map), B=4 N=4096 H=8 D=M=64, fp32 in/out.
// HBM-bound: ~128MB traffic @ 23.3TB/s => ~5.5us floor; ~2.1 GFLOP is trivial
// for bf16 WMMA. Design: read each input once, coalesced; overlap global->LDS
// movement with WMMA using CDNA5 TDM (TENSORcnt) / async loads (ASYNCcnt).
//   phase 1: KV[m,d] = sum_n V[n,m] * phi(K[n,d]),  Ksum[d] = sum_n phi(K[n,d])
//   phase 2: Out[n,m] = (phi(Q[n,:]) @ KV^T) * 1/(phi(Q[n,:]).Ksum + eps)
// ---------------------------------------------------------------------------

typedef __attribute__((ext_vector_type(16))) __bf16 v16bf;
typedef __attribute__((ext_vector_type(8)))  float  v8f;
typedef __attribute__((ext_vector_type(4)))  unsigned int u32x4;
typedef __attribute__((ext_vector_type(8)))  int    i32x8;
typedef __attribute__((ext_vector_type(4)))  int    i32x4;
typedef __attribute__((ext_vector_type(4)))  int    v4i;

// CDNA5 data-mover paths (guarded: fall back to sync staging if unavailable)
#if defined(__AMDGCN__) && __has_builtin(__builtin_amdgcn_tensor_load_to_lds) && \
    __has_builtin(__builtin_amdgcn_s_wait_tensorcnt)
#define USE_TDM 1
#else
#define USE_TDM 0
#endif

#if defined(__AMDGCN__) && __has_builtin(__builtin_amdgcn_global_load_async_to_lds_b128) && \
    __has_builtin(__builtin_amdgcn_s_wait_asynccnt)
#define USE_ASYNC 1
#else
#define USE_ASYNC 0
#endif

#define B_      4
#define N_      4096
#define H_      8
#define D_      64
#define M_      64
#define BH      (B_ * H_)          // 32
#define CH      8                  // N-chunks for phase-1 partials
#define CHROWS  (N_ / CH)          // 512 rows per chunk
#define ROWSTEP 32                 // rows staged per LDS tile (= WMMA K)
#define NTILE   128                // output rows per phase-2 workgroup
#define NTILES  (N_ / NTILE)       // 32
#define EPS_    1e-6f
#define LP      72                 // LDS row pitch (bf16) to dodge bank conflicts

// workspace layout (floats)
#define WS_KV(bh, c)  ((size_t)((bh) * CH + (c)) * (M_ * D_))
#define WS_KS_BASE    ((size_t)BH * CH * M_ * D_)
#define WS_KS(bh, c)  (WS_KS_BASE + (size_t)((bh) * CH + (c)) * D_)

__device__ __forceinline__ float fmap(float x) {
  return x > 0.0f ? x + 1.0f : __expf(x);   // elu(x)+1
}

// 16-bit A-matrix 16x32 per-lane K index (ISA 7.12.2)
__device__ __forceinline__ int a_kidx(int e, int lane) {
  return ((e >> 3) << 4) + ((lane >> 4) << 3) + (e & 7);
}

#if USE_TDM
// Issue one TDM descriptor: load a [rows x 64] f32 tile whose rows are
// strided by H_*D_ elements from global into LDS (CDNA5 ISA).
// amdgpu-toolchain builtin arity: (u32x4, i32x8, i32x4, i32x4, i32x8, i32).
__device__ __forceinline__ void tdm_load_tile(const float* gsrc, void* ldsdst,
                                              int rows) {
  unsigned long long ga = (unsigned long long)(uintptr_t)gsrc;
  unsigned lds = (unsigned)(uintptr_t)ldsdst;
  u32x4 g0;
  g0[0] = 1u;                                         // count=1 (valid user D#)
  g0[1] = lds;                                        // lds_addr
  g0[2] = (unsigned)(ga & 0xffffffffu);               // global_addr[31:0]
  g0[3] = (unsigned)((ga >> 32) & 0x01ffffffu) | (2u << 30);  // [56:32], type=2
  i32x8 g1;
  g1[0] = (int)(2u << 16);                  // wg_mask=0, data_size=2 (4 bytes)
  g1[1] = (int)((unsigned)D_ << 16);        // tensor_dim0[15:0] = 64
  g1[2] = (int)(((unsigned)N_ & 0xffffu) << 16);  // dim0[31:16]=0 | dim1[15:0]
  g1[3] = (int)((unsigned)D_ << 16);        // dim1[31:16]=0 | tile_dim0 = 64
  g1[4] = (int)(rows & 0xffff);             // tile_dim1 = rows | tile_dim2 = 0
  g1[5] = (int)(H_ * D_);                   // tensor_dim0_stride[31:0] = 512
  g1[6] = 0;                                // stride0[47:32]=0 | stride1[15:0]
  g1[7] = 0;
  i32x4 z4 = {0, 0, 0, 0};                  // groups 2/3 unused (2D tensor)
  i32x8 z8 = {0, 0, 0, 0, 0, 0, 0, 0};
  __builtin_amdgcn_tensor_load_to_lds(g0, g1, z4, z4, z8, 0);
}
#endif

#if USE_ASYNC
__device__ __forceinline__ void async_cp16(const float* g, float* l) {
  __builtin_amdgcn_global_load_async_to_lds_b128(
      (__attribute__((address_space(1))) v4i*)(uintptr_t)g,
      (__attribute__((address_space(3))) v4i*)(unsigned)(uintptr_t)l, 0, 0);
}
#endif

// ---------------------------------------------------------------------------
// Kernel 1: per (bh, chunk) partial KV (64x64) and partial Ksum (64).
// TDM double-buffers the 32x64 f32 K/V tiles into LDS; VALU applies phi and
// converts to bf16; WMMA contracts over n in steps of 32.
// ---------------------------------------------------------------------------
__global__ __launch_bounds__(256) void kv_partial_kernel(
    const float* __restrict__ Kin, const float* __restrict__ Vin,
    float* __restrict__ ws) {
#if USE_TDM
  __shared__ float Ksrc[2][ROWSTEP][D_];
  __shared__ float Vsrc[2][ROWSTEP][D_];
#endif
  __shared__ __bf16 Kt[ROWSTEP][LP];
  __shared__ __bf16 Vt[ROWSTEP][LP];
  __shared__ float  red[ROWSTEP][D_];

  const int tid  = threadIdx.x;
  const int lane = tid & 31;
  const int w    = tid >> 5;
  const int bh   = blockIdx.x / CH;
  const int c    = blockIdx.x % CH;
  const int b    = bh / H_;
  const int h    = bh % H_;
  const int n0   = c * CHROWS;

  const int nloc = tid >> 3;        // 0..31: staged row
  const int d0   = (tid & 7) * 8;   // 8 consecutive feature elements

  const int mt  = w >> 1;           // KV row tile (m)
  const int dtb = (w & 1) * 2;      // KV col tile pair (d)
  v8f acc0 = {};
  v8f acc1 = {};

  float ksum8[8];
#pragma unroll
  for (int j = 0; j < 8; ++j) ksum8[j] = 0.0f;

  const int    NIT       = CHROWS / ROWSTEP;
  const size_t tilebase  = ((size_t)(b * N_ + n0) * H_ + h) * (size_t)D_;
  const size_t tilestep  = (size_t)ROWSTEP * H_ * D_;

#if USE_TDM
  if (w == 0) {  // wave 0 drives the tensor data mover
    tdm_load_tile(Kin + tilebase, &Ksrc[0][0][0], ROWSTEP);
    tdm_load_tile(Vin + tilebase, &Vsrc[0][0][0], ROWSTEP);
  }
#endif

  for (int it = 0; it < NIT; ++it) {
    float kf[8], vf[8];
#if USE_TDM
    if (w == 0) {
      if (it + 1 < NIT) {   // prefetch next slab into the other buffer
        const size_t nb = tilebase + (size_t)(it + 1) * tilestep;
        tdm_load_tile(Kin + nb, &Ksrc[(it + 1) & 1][0][0], ROWSTEP);
        tdm_load_tile(Vin + nb, &Vsrc[(it + 1) & 1][0][0], ROWSTEP);
        __builtin_amdgcn_s_wait_tensorcnt(2);  // current slab landed
      } else {
        __builtin_amdgcn_s_wait_tensorcnt(0);
      }
    }
    __syncthreads();
    {
      const float* kp = &Ksrc[it & 1][nloc][d0];
      const float* vp = &Vsrc[it & 1][nloc][d0];
#pragma unroll
      for (int j = 0; j < 8; ++j) {
        kf[j] = fmap(kp[j]);
        vf[j] = vp[j];
      }
    }
#else
    {
      const int n = n0 + it * ROWSTEP + nloc;
      const size_t base = ((size_t)(b * N_ + n) * H_ + h) * (size_t)D_ + d0;
      const float4 k0 = *(const float4*)(Kin + base);
      const float4 k1 = *(const float4*)(Kin + base + 4);
      const float4 v0 = *(const float4*)(Vin + base);
      const float4 v1 = *(const float4*)(Vin + base + 4);
      if (it + 1 < NIT) {
        __builtin_prefetch(Kin + base + tilestep, 0, 1);
        __builtin_prefetch(Vin + base + tilestep, 0, 1);
      }
      kf[0] = fmap(k0.x); kf[1] = fmap(k0.y); kf[2] = fmap(k0.z); kf[3] = fmap(k0.w);
      kf[4] = fmap(k1.x); kf[5] = fmap(k1.y); kf[6] = fmap(k1.z); kf[7] = fmap(k1.w);
      vf[0] = v0.x; vf[1] = v0.y; vf[2] = v0.z; vf[3] = v0.w;
      vf[4] = v1.x; vf[5] = v1.y; vf[6] = v1.z; vf[7] = v1.w;
    }
#endif
#pragma unroll
    for (int j = 0; j < 8; ++j) {
      Kt[nloc][d0 + j] = (__bf16)kf[j];
      Vt[nloc][d0 + j] = (__bf16)vf[j];
      ksum8[j] += kf[j];
    }
    __syncthreads();

    // WMMA: A[m,n'] = V[n',m], B[n',d] = phi(K)[n',d]
    v16bf a;
#pragma unroll
    for (int e = 0; e < 16; ++e)
      a[e] = Vt[a_kidx(e, lane)][mt * 16 + (lane & 15)];
    v16bf fb0, fb1;
#pragma unroll
    for (int e = 0; e < 16; ++e) {
      const int kr = a_kidx(e, lane);
      fb0[e] = Kt[kr][(dtb + 0) * 16 + (lane & 15)];
      fb1[e] = Kt[kr][(dtb + 1) * 16 + (lane & 15)];
    }
    acc0 = __builtin_amdgcn_wmma_f32_16x16x32_bf16(false, a, false, fb0,
                                                   (short)0, acc0, false, false);
    acc1 = __builtin_amdgcn_wmma_f32_16x16x32_bf16(false, a, false, fb1,
                                                   (short)0, acc1, false, false);
    __syncthreads();
  }

  // write partial KV: C layout => row m = mt*16 + (lane>=16?8:0) + r
  float* kvp = ws + WS_KV(bh, c);
#pragma unroll
  for (int r = 0; r < 8; ++r) {
    const int m = mt * 16 + (lane >> 4) * 8 + r;
    kvp[m * D_ + (dtb + 0) * 16 + (lane & 15)] = acc0[r];
    kvp[m * D_ + (dtb + 1) * 16 + (lane & 15)] = acc1[r];
  }

  // reduce per-thread Ksum partials across the 32 row-slots
#pragma unroll
  for (int j = 0; j < 8; ++j) red[nloc][d0 + j] = ksum8[j];
  __syncthreads();
  if (tid < D_) {
    float s = 0.0f;
    for (int r = 0; r < ROWSTEP; ++r) s += red[r][tid];
    ws[WS_KS(bh, c) + tid] = s;
  }
}

// ---------------------------------------------------------------------------
// Kernel 2: reduce partials, then Out[n,m] = (phi(Q) @ KV^T) * Z per row-tile.
// Async global->LDS (ASYNCcnt) double-buffers the Q tiles.
// ---------------------------------------------------------------------------
__global__ __launch_bounds__(256) void attn_out_kernel(
    const float* __restrict__ Qin, const float* __restrict__ ws,
    float* __restrict__ Out) {
#if USE_ASYNC
  __shared__ float Qsrc[2][ROWSTEP][D_];
#endif
  __shared__ __bf16 KVs[D_][LP];   // [d][m] == B-matrix natural layout
  __shared__ float  Ks[D_];
  __shared__ __bf16 Qt[ROWSTEP][LP];
  __shared__ float  Zl[ROWSTEP];

  const int tid  = threadIdx.x;
  const int lane = tid & 31;
  const int w    = tid >> 5;
  const int bh   = blockIdx.x / NTILES;
  const int nt   = blockIdx.x % NTILES;
  const int b    = bh / H_;
  const int h    = bh % H_;
  const int n0   = nt * NTILE;

  const int nloc = tid >> 3;
  const int d0   = (tid & 7) * 8;
  const int NP   = NTILE / ROWSTEP;
  const size_t tilebase = ((size_t)(b * N_ + n0 + nloc) * H_ + h) * (size_t)D_ + d0;
  const size_t tilestep = (size_t)ROWSTEP * H_ * D_;

#if USE_ASYNC
  {  // kick off pass-0 Q tile while we reduce KV partials below
    const float* g = Qin + tilebase;
    async_cp16(g,     &Qsrc[0][nloc][d0]);
    async_cp16(g + 4, &Qsrc[0][nloc][d0 + 4]);
  }
#endif

  // ---- reduce KV partials: thread t owns m = t>>2, 16 d's ----
  {
    const int m  = tid >> 2;
    const int db = (tid & 3) * 16;
    float acc[16];
#pragma unroll
    for (int j = 0; j < 16; ++j) acc[j] = 0.0f;
    for (int c = 0; c < CH; ++c) {
      const float* pp = ws + WS_KV(bh, c) + m * D_ + db;
#pragma unroll
      for (int q4 = 0; q4 < 4; ++q4) {
        const float4 v = *(const float4*)(pp + q4 * 4);
        acc[q4 * 4 + 0] += v.x; acc[q4 * 4 + 1] += v.y;
        acc[q4 * 4 + 2] += v.z; acc[q4 * 4 + 3] += v.w;
      }
    }
#pragma unroll
    for (int j = 0; j < 16; ++j) KVs[db + j][m] = (__bf16)acc[j];
    if (tid < D_) {
      float s = 0.0f;
      for (int c = 0; c < CH; ++c) s += ws[WS_KS(bh, c) + tid];
      Ks[tid] = s;
    }
  }
  __syncthreads();

  const int rt = w >> 2;   // row tile within the 32-row pass (0..1)
  const int ct = w & 3;    // column tile (0..3)

  for (int p = 0; p < NP; ++p) {
    float qf[8];
#if USE_ASYNC
    if (p + 1 < NP) {      // prefetch next Q tile into the other buffer
      const float* g = Qin + tilebase + (size_t)(p + 1) * tilestep;
      async_cp16(g,     &Qsrc[(p + 1) & 1][nloc][d0]);
      async_cp16(g + 4, &Qsrc[(p + 1) & 1][nloc][d0 + 4]);
      __builtin_amdgcn_s_wait_asynccnt(2);   // current tile landed
    } else {
      __builtin_amdgcn_s_wait_asynccnt(0);
    }
    __syncthreads();
    {
      const float* qp = &Qsrc[p & 1][nloc][d0];
#pragma unroll
      for (int j = 0; j < 8; ++j) qf[j] = fmap(qp[j]);
    }
#else
    {
      const size_t base = tilebase + (size_t)p * tilestep;
      const float4 q0 = *(const float4*)(Qin + base);
      const float4 q1 = *(const float4*)(Qin + base + 4);
      if (p + 1 < NP) __builtin_prefetch(Qin + base + tilestep, 0, 1);
      qf[0] = fmap(q0.x); qf[1] = fmap(q0.y); qf[2] = fmap(q0.z); qf[3] = fmap(q0.w);
      qf[4] = fmap(q1.x); qf[5] = fmap(q1.y); qf[6] = fmap(q1.z); qf[7] = fmap(q1.w);
    }
#endif
#pragma unroll
    for (int j = 0; j < 8; ++j) Qt[nloc][d0 + j] = (__bf16)qf[j];
    __syncthreads();

    // Z = 1/(phi(Q).Ksum + eps) for the 32 staged rows
    if (tid < ROWSTEP) {
      float s = 0.0f;
      for (int d = 0; d < D_; ++d) s += (float)Qt[tid][d] * Ks[d];
      Zl[tid] = 1.0f / (s + EPS_);
    }
    __syncthreads();

    // Out tile = phi(Q)(16x64) @ KV^T(64x16) via 2 WMMA k-steps
    v8f acc = {};
#pragma unroll
    for (int s = 0; s < 2; ++s) {
      const int kb = s * 32;
      v16bf a, fb;
#pragma unroll
      for (int e = 0; e < 16; ++e) {
        a[e]  = Qt[rt * 16 + (lane & 15)][kb + a_kidx(e, lane)];
        fb[e] = KVs[kb + a_kidx(e, lane)][ct * 16 + (lane & 15)];
      }
      acc = __builtin_amdgcn_wmma_f32_16x16x32_bf16(false, a, false, fb,
                                                    (short)0, acc, false, false);
    }

    // scale by Z and store
#pragma unroll
    for (int r = 0; r < 8; ++r) {
      const int nl = rt * 16 + (lane >> 4) * 8 + r;
      const int nn = n0 + p * ROWSTEP + nl;
      const int m  = ct * 16 + (lane & 15);
      Out[((size_t)(b * N_ + nn) * H_ + h) * (size_t)M_ + m] = acc[r] * Zl[nl];
    }
    __syncthreads();
  }
}

extern "C" void kernel_launch(void* const* d_in, const int* in_sizes, int n_in,
                              void* d_out, int out_size, void* d_ws,
                              size_t ws_size, hipStream_t stream) {
  (void)in_sizes; (void)n_in; (void)out_size; (void)ws_size;
  const float* Q = (const float*)d_in[0];
  const float* K = (const float*)d_in[1];
  const float* V = (const float*)d_in[2];
  // d_in[3] = attn_mask (unused by the reference math)
  float* Out = (float*)d_out;
  float* ws  = (float*)d_ws;  // ~4.1 MB: 32*8*(64*64 + 64) floats

  kv_partial_kernel<<<BH * CH, 256, 0, stream>>>(K, V, ws);
  attn_out_kernel<<<BH * NTILES, 256, 0, stream>>>(Q, ws, Out);
}
